// LDPCModel_59545426592349
// MI455X (gfx1250) — matrix operand order repaired and checked by reading, not compiled, not run on hardware
//
#include <hip/hip_runtime.h>

typedef __attribute__((ext_vector_type(16))) _Float16 v16h;
typedef __attribute__((ext_vector_type(8)))  _Float16 v8h;
typedef __attribute__((ext_vector_type(8)))  float    v8f;

#define NWAVES 8

struct KP { const void* p[128]; };

#define PF(i) ((const float*)P.p[(i)])
#define PI(i) ((const int*)P.p[(i)])

// ---- d_in leaf indices (inputs in setup order, params as jax tree_leaves:
// dict keys sorted alphabetically, lists in order) ----
enum {
  I_NODE = 0, I_HOP = 1, I_IDX_F2V = 2, I_IDX_V2F = 3, I_EF_F2V = 4, I_EF_V2F = 5,
  P_BN_B = 6, P_BN_G = 7,
  P_EMF2V_B1 = 8, P_EMF2V_B2 = 9, P_EMF2V_W1 = 10, P_EMF2V_W2 = 11,
  P_EMV2F_B1 = 12, P_EMV2F_B2 = 13, P_EMV2F_W1 = 14, P_EMV2F_W2 = 15,
  P_FEMB_B0 = 16, P_FEMB_B1 = 17, P_FEMB_W0 = 18, P_FEMB_W1 = 19,
  P_LAYERS = 20,   // 14 leaves per layer x 7 layers = 20..117
  P_OUT_B = 118, P_OUT_W = 119,
  P_REG_B1 = 120, P_REG_B2 = 121, P_REG_B3 = 122,
  P_REG_W1 = 123, P_REG_W2 = 124, P_REG_W3 = 125,
  P_VEMB_B = 126, P_VEMB_W = 127
};
// per-layer leaf order (sorted keys: f2v_W, f2v_b, fup_b, fup_w, v2f_W, v2f_b, vup_b, vup_w)
enum { L_F2V_W0 = 0, L_F2V_W1, L_F2V_B0, L_F2V_B1, L_FUP_B0, L_FUP_B1,
       L_FUP_W0, L_FUP_W1, L_V2F_W0, L_V2F_W1, L_V2F_B0, L_V2F_B1,
       L_VUP_B, L_VUP_W };

// ---- WMMA fragment loaders (contiguous k order, same for A and B) ----
// A 16x32: lane (m=L&15, half=L>>4); element e -> K = 16*half + e.
//          One contiguous f32 run (matches B).
__device__ __forceinline__ v16h load_a(const float* wrow, int half) {
  v16h a;
#pragma unroll
  for (int e = 0; e < 8; ++e) {
    a[e]     = (_Float16)wrow[16 * half + e];
    a[8 + e] = (_Float16)wrow[16 * half + 8 + e];
  }
  return a;
}

// B 32x16: lane (n=L&15, half=L>>4); element e -> K = 16*half + e.
// With channel-contiguous LDS rows this is 16 consecutive halves (2x b128).
__device__ __forceinline__ v16h load_b16(const _Float16* p) {
  const v8h* q = (const v8h*)p;
  v8h b0 = q[0], b1 = q[1];
  v16h r;
#pragma unroll
  for (int e = 0; e < 8; ++e) { r[e] = b0[e]; r[8 + e] = b1[e]; }
  return r;
}

__device__ __forceinline__ v8f wmma_dense(v8f c, const float* wrow,
                                          const _Float16* brow, int half) {
  v16h a = load_a(wrow, half);
  v16h b = load_b16(brow + half * 16);
  return __builtin_amdgcn_wmma_f32_16x16x32_f16(false, a, false, b, (short)0, c,
                                                false, false);
}

__device__ __forceinline__ v8f wmma_gather(v8f c, const float* wrow,
                                           const _Float16* grow, _Float16 eth,
                                           int half) {
  v16h a = load_a(wrow, half);
  v16h g = load_b16(grow + half * 16);
  v16h b;
#pragma unroll
  for (int e = 0; e < 16; ++e) b[e] = eth * g[e];  // v_pk_mul_f16 pairs
  return __builtin_amdgcn_wmma_f32_16x16x32_f16(false, a, false, b, (short)0, c,
                                                false, false);
}

__global__ __launch_bounds__(256)
void ldpc_main(KP P, float* __restrict__ bits_out, float* __restrict__ f1_out) {
  // ---- LDS (~155 KB), all activation buffers node-major / channel-contiguous ----
  __shared__ _Float16 sh_vin[96 * 320];     // row n: concat(vin, msg) channels
  __shared__ _Float16 sh_V[96 * 128];       // row n: V channels; aliased as f0new
  __shared__ _Float16 sh_saved[3 * 96 * 64];
  __shared__ _Float16 sh_fcat[48 * 256];    // row n: concat(F0, mv0) channels
  __shared__ float sh_et_f2v[3 * 288];
  __shared__ float sh_et_v2f[3 * 288];
  __shared__ short sh_idx_f2v[288];
  __shared__ short sh_idx_v2f[288];
  __shared__ float sh_nf0[96];
  __shared__ float sh_F1[128];
  __shared__ float sh_mv1[128];
  __shared__ float sh_F1new[128];
  __shared__ float sh_w1f1[128];

  _Float16* sh_f0new = sh_V;  // alias: sh_V dead between vin-build and vup (48x128 rows)

  const int tid  = threadIdx.x;
  const int wv   = tid >> 5;
  const int lane = tid & 31;
  const int half = lane >> 4;
  const int lm   = lane & 15;
  const int b    = blockIdx.x;

  constexpr int FIL[7]   = {64, 64, 64, 128, 128, 64, 64};
  constexpr int SKIPS[7] = {-1, -1, -1, 2, 1, 0, -1};

  // ---- gather indices + node-feature channel 0 ----
  for (int i = tid; i < 288; i += 256) {
    sh_idx_f2v[i] = (short)PI(I_IDX_F2V)[b * 288 + i];
    sh_idx_v2f[i] = (short)PI(I_IDX_V2F)[b * 288 + i];
  }
  for (int n = tid; n < 96; n += 256)
    sh_nf0[n] = PF(I_NODE)[(size_t)b * 768 + n];
  __syncthreads();

  // ---- edge-type embedding MLPs: 7 -> 64 -> 3 per edge slot ----
  {
    const float* W1 = PF(P_EMF2V_W1); const float* B1 = PF(P_EMF2V_B1);
    const float* W2 = PF(P_EMF2V_W2); const float* B2 = PF(P_EMF2V_B2);
    const float* ef = PF(I_EF_F2V) + (size_t)b * 7 * 288;
    for (int p = tid; p < 288; p += 256) {
      float x[7];
#pragma unroll
      for (int c = 0; c < 7; ++c) x[c] = ef[c * 288 + p];
      float a0 = B2[0], a1 = B2[1], a2 = B2[2];
      for (int j = 0; j < 64; ++j) {
        float h = B1[j];
#pragma unroll
        for (int c = 0; c < 7; ++c) h += W1[j * 7 + c] * x[c];
        h = fmaxf(h, 0.f);
        a0 += W2[j] * h; a1 += W2[64 + j] * h; a2 += W2[128 + j] * h;
      }
      sh_et_f2v[p] = a0; sh_et_f2v[288 + p] = a1; sh_et_f2v[576 + p] = a2;
    }
  }
  {
    const float* W1 = PF(P_EMV2F_W1); const float* B1 = PF(P_EMV2F_B1);
    const float* W2 = PF(P_EMV2F_W2); const float* B2 = PF(P_EMV2F_B2);
    const float* ef = PF(I_EF_V2F) + (size_t)b * 7 * 288;
    for (int p = tid; p < 288; p += 256) {
      float x[7];
#pragma unroll
      for (int c = 0; c < 7; ++c) x[c] = ef[c * 288 + p];
      float a0 = B2[0], a1 = B2[1], a2 = B2[2];
      for (int j = 0; j < 64; ++j) {
        float h = B1[j];
#pragma unroll
        for (int c = 0; c < 7; ++c) h += W1[j * 7 + c] * x[c];
        h = fmaxf(h, 0.f);
        a0 += W2[j] * h; a1 += W2[64 + j] * h; a2 += W2[128 + j] * h;
      }
      sh_et_v2f[p] = a0; sh_et_v2f[288 + p] = a1; sh_et_v2f[576 + p] = a2;
    }
  }

  // ---- initial embeddings ----
  {  // V = relu(Wv(64x8) @ node_feature)  -> sh_V[n*128 + o]
    const float* w = PF(P_VEMB_W); const float* bb = PF(P_VEMB_B);
    const float* nf = PF(I_NODE) + (size_t)b * 768;
    for (int i = tid; i < 96 * 64; i += 256) {
      int n = i >> 6, o = i & 63;
      float acc = bb[o];
#pragma unroll
      for (int c = 0; c < 8; ++c) acc += w[o * 8 + c] * nf[c * 96 + n];
      sh_V[n * 128 + o] = (_Float16)fmaxf(acc, 0.f);
    }
  }
  {  // F0 = relu(Wf0(64x6) @ hop)  -> sh_fcat[n*256 + o]
    const float* w = PF(P_FEMB_W0); const float* bb = PF(P_FEMB_B0);
    const float* hp = PF(I_HOP) + (size_t)b * 288;
    for (int i = tid; i < 48 * 64; i += 256) {
      int n = i >> 6, o = i & 63;
      float acc = bb[o];
#pragma unroll
      for (int c = 0; c < 6; ++c) acc += w[o * 6 + c] * hp[c * 48 + n];
      sh_fcat[n * 256 + o] = (_Float16)fmaxf(acc, 0.f);
    }
  }
  {  // F1 = relu(Wf1(64x96) @ nhop)
    const float* w = PF(P_FEMB_W1); const float* bb = PF(P_FEMB_B1);
    for (int o = tid; o < 64; o += 256) {
      float acc = bb[o];
      for (int n = 0; n < 96; ++n) acc += w[o * 96 + n] * sh_nf0[n];
      sh_F1[o] = fmaxf(acc, 0.f);
    }
  }
  __syncthreads();

  int ch_v = 64, chf = 64;
  for (int l = 0; l < 7; ++l) {
    const int cout = FIL[l];
    const int skip = SKIPS[l];
    const int cinv = ch_v + ((skip >= 0) ? 64 : 0);
    const int lb   = P_LAYERS + 14 * l;
    const int MT   = cout >> 4;

    if (lane == 0) {  // warm caches for the layer's big weights
      __builtin_prefetch(PF(lb + L_V2F_W0));
      __builtin_prefetch(PF(lb + L_F2V_W0));
      __builtin_prefetch(PF(lb + L_VUP_W));
    }

    // ---- build vin rows: [V | saved[skip]] ----
    for (int i = tid; i < 96 * ch_v; i += 256) {
      int n = i / ch_v, c = i - n * ch_v;   // ch_v is 64 or 128
      sh_vin[n * 320 + c] = sh_V[n * 128 + c];
    }
    if (skip >= 0)
      for (int i = tid; i < 96 * 64; i += 256) {
        int n = i >> 6, c = i & 63;
        sh_vin[n * 320 + ch_v + c] = sh_saved[skip * 96 * 64 + n * 64 + c];
      }
    __syncthreads();

    // ---- v2f g=0: mv0[o,n] = max_k( et.W @ gather(vin) + et.b ) ----
    {
      const float* W0 = PF(lb + L_V2F_W0);
      const float* B0 = PF(lb + L_V2F_B0);
      const int KT = (3 * cinv) >> 5;
      for (int tile = wv; tile < MT * 3; tile += NWAVES) {
        const int mo = tile / 3, no = tile % 3;
        const int node = no * 16 + lm;
        v8f amax;
#pragma unroll
        for (int r = 0; r < 8; ++r) amax[r] = -3.4e38f;
        for (int ke = 0; ke < 6; ++ke) {
          const int p = node * 6 + ke;
          const int src = sh_idx_v2f[p];
          const float e0 = sh_et_v2f[p], e1 = sh_et_v2f[288 + p], e2 = sh_et_v2f[576 + p];
          v8f c = {};
          for (int kt = 0; kt < KT; ++kt) {
            const int kk0 = kt << 5;
            const int t = kk0 / cinv;
            const int cb = kk0 - t * cinv;
            const _Float16 eth = (_Float16)((t == 0) ? e0 : (t == 1) ? e1 : e2);
            const float* wrow = W0 + (size_t)(t * cout + mo * 16 + lm) * cinv + cb;
            c = wmma_gather(c, wrow, sh_vin + src * 320 + cb, eth, half);
          }
#pragma unroll
          for (int r = 0; r < 8; ++r) {
            const int o = mo * 16 + r + 8 * half;
            const float bias = e0 * B0[o] + e1 * B0[cout + o] + e2 * B0[2 * cout + o];
            amax[r] = fmaxf(amax[r], c[r] + bias);
          }
        }
        v8h outv;
#pragma unroll
        for (int r = 0; r < 8; ++r) outv[r] = (_Float16)amax[r];
        *(v8h*)&sh_fcat[node * 256 + chf + mo * 16 + 8 * half] = outv;
      }
    }
    __syncthreads();

    // ---- v2f g=1: mv1[o] = max_n( W1 @ vin[:,n] ) + b1[o] ----
    {
      const float* W1 = PF(lb + L_V2F_W1);
      const float* B1 = PF(lb + L_V2F_B1);
      const int KT = cinv >> 5;
      if (wv < MT) {
        const int mo = wv;
        v8f amax;
#pragma unroll
        for (int r = 0; r < 8; ++r) amax[r] = -3.4e38f;
        for (int no = 0; no < 6; ++no) {
          v8f c = {};
          for (int kt = 0; kt < KT; ++kt) {
            const float* wrow = W1 + (size_t)(mo * 16 + lm) * cinv + (kt << 5);
            c = wmma_dense(c, wrow, sh_vin + (no * 16 + lm) * 320 + (kt << 5), half);
          }
#pragma unroll
          for (int r = 0; r < 8; ++r) amax[r] = fmaxf(amax[r], c[r]);
        }
#pragma unroll
        for (int r = 0; r < 8; ++r) {
          float v = amax[r];
          for (int s = 1; s < 16; s <<= 1) v = fmaxf(v, __shfl_xor(v, s, 32));
          if (lm == 0) {
            const int o = mo * 16 + r + 8 * half;
            sh_mv1[o] = v + B1[o];
          }
        }
      }
    }
    __syncthreads();

    // ---- fup g=0: F0new = relu(Wfup0 @ concat(F0, mv0)) ----
    {
      const float* Wf = PF(lb + L_FUP_W0);
      const float* bf = PF(lb + L_FUP_B0);
      const int Kdim = chf + cout;
      const int KT = Kdim >> 5;
      for (int tile = wv; tile < MT * 3; tile += NWAVES) {
        const int mo = tile / 3, no = tile % 3;
        const int node = no * 16 + lm;
        v8f c = {};
        for (int kt = 0; kt < KT; ++kt) {
          const float* wrow = Wf + (size_t)(mo * 16 + lm) * Kdim + (kt << 5);
          c = wmma_dense(c, wrow, sh_fcat + node * 256 + (kt << 5), half);
        }
        v8h outv;
#pragma unroll
        for (int r = 0; r < 8; ++r) {
          const int o = mo * 16 + r + 8 * half;
          outv[r] = (_Float16)fmaxf(c[r] + bf[o], 0.f);
        }
        *(v8h*)&sh_f0new[node * 128 + mo * 16 + 8 * half] = outv;
      }
    }
    __syncthreads();

    // ---- fup g=1 (matvec) ----
    {
      const float* Wf = PF(lb + L_FUP_W1);
      const float* bf = PF(lb + L_FUP_B1);
      const int Kdim = chf + cout;
      for (int o = tid; o < cout; o += 256) {
        float acc = bf[o];
        const float* wr = Wf + (size_t)o * Kdim;
        for (int c0 = 0; c0 < chf; ++c0) acc += wr[c0] * sh_F1[c0];
        for (int c0 = 0; c0 < cout; ++c0) acc += wr[chf + c0] * sh_mv1[c0];
        sh_F1new[o] = fmaxf(acc, 0.f);
      }
    }
    __syncthreads();

    // ---- f2v g=1: broadcast message w1f1 = W @ F1new + b ----
    {
      const float* W1 = PF(lb + L_F2V_W1);
      const float* B1 = PF(lb + L_F2V_B1);
      for (int o = tid; o < cout; o += 256) {
        float acc = B1[o];
        const float* wr = W1 + (size_t)o * cout;
        for (int c0 = 0; c0 < cout; ++c0) acc += wr[c0] * sh_F1new[c0];
        sh_w1f1[o] = acc;
      }
    }
    __syncthreads();

    // ---- f2v g=0: msg[o,n] = max_k( et.W @ gather(F0new) + et.b ) + w1f1[o] ----
    {
      const float* W0 = PF(lb + L_F2V_W0);
      const float* B0 = PF(lb + L_F2V_B0);
      const int KT = (3 * cout) >> 5;
      for (int tile = wv; tile < MT * 6; tile += NWAVES) {
        const int mo = tile / 6, no = tile % 6;
        const int node = no * 16 + lm;
        v8f amax;
#pragma unroll
        for (int r = 0; r < 8; ++r) amax[r] = -3.4e38f;
        for (int ke = 0; ke < 3; ++ke) {
          const int p = node * 3 + ke;
          const int src = sh_idx_f2v[p];
          const float e0 = sh_et_f2v[p], e1 = sh_et_f2v[288 + p], e2 = sh_et_f2v[576 + p];
          v8f c = {};
          for (int kt = 0; kt < KT; ++kt) {
            const int kk0 = kt << 5;
            const int t = kk0 / cout;
            const int cb = kk0 - t * cout;
            const _Float16 eth = (_Float16)((t == 0) ? e0 : (t == 1) ? e1 : e2);
            const float* wrow = W0 + (size_t)(t * cout + mo * 16 + lm) * cout + cb;
            c = wmma_gather(c, wrow, sh_f0new + src * 128 + cb, eth, half);
          }
#pragma unroll
          for (int r = 0; r < 8; ++r) {
            const int o = mo * 16 + r + 8 * half;
            const float bias = e0 * B0[o] + e1 * B0[cout + o] + e2 * B0[2 * cout + o];
            amax[r] = fmaxf(amax[r], c[r] + bias);
          }
        }
        v8h outv;
#pragma unroll
        for (int r = 0; r < 8; ++r) {
          const int o = mo * 16 + r + 8 * half;
          outv[r] = (_Float16)(amax[r] + sh_w1f1[o]);
        }
        *(v8h*)&sh_vin[node * 320 + cinv + mo * 16 + 8 * half] = outv;
      }
    }
    __syncthreads();

    // ---- carry F state forward (before vup clobbers the f0new alias) ----
    for (int o = tid; o < cout; o += 256) sh_F1[o] = sh_F1new[o];
    for (int i = tid; i < 48 * cout; i += 256) {
      int n = i / cout, o = i - n * cout;   // cout is 64 or 128
      sh_fcat[n * 256 + o] = sh_f0new[n * 128 + o];
    }
    __syncthreads();

    // ---- vup: V = relu(Wvup @ concat(vin, msg)) ----
    {
      const float* Wv = PF(lb + L_VUP_W);
      const float* bv = PF(lb + L_VUP_B);
      const int Kdim = cinv + cout;
      const int KT = Kdim >> 5;
      for (int tile = wv; tile < MT * 6; tile += NWAVES) {
        const int mo = tile / 6, no = tile % 6;
        const int n = no * 16 + lm;
        v8f c = {};
        for (int kt = 0; kt < KT; ++kt) {
          const float* wrow = Wv + (size_t)(mo * 16 + lm) * Kdim + (kt << 5);
          c = wmma_dense(c, wrow, sh_vin + n * 320 + (kt << 5), half);
        }
        v8h outv;
#pragma unroll
        for (int r = 0; r < 8; ++r) {
          const int o = mo * 16 + r + 8 * half;
          outv[r] = (_Float16)fmaxf(c[r] + bv[o], 0.f);
        }
        *(v8h*)&sh_V[n * 128 + mo * 16 + 8 * half] = outv;
        if (l <= 2)
          *(v8h*)&sh_saved[l * 96 * 64 + n * 64 + mo * 16 + 8 * half] = outv;
      }
    }
    __syncthreads();

    ch_v = cout;
    chf  = cout;
  }

  // ---- bits = out_w @ V + out_b + node_feature[ch0], first 48 vars ----
  {
    const float* ow = PF(P_OUT_W);
    const float ob = PF(P_OUT_B)[0];
    for (int n = tid; n < 48; n += 256) {
      float acc = ob + sh_nf0[n];
      for (int c = 0; c < 64; ++c) acc += ow[c] * (float)sh_V[n * 128 + c];
      bits_out[b * 48 + n] = acc;
    }
  }
  // ---- export F1 for regression head ----
  for (int o = tid; o < 64; o += 256) f1_out[b * 64 + o] = sh_F1[o];
}

// ==================== regression head (cross-batch batchnorm) ====================

__global__ __launch_bounds__(128)
void ldpc_head1(KP P, const float* __restrict__ f1, float* __restrict__ h1) {
  const int b = blockIdx.x, j = threadIdx.x;
  const float* w = PF(P_REG_W1);
  float acc = PF(P_REG_B1)[j];
  const float* f = f1 + b * 64;
  for (int c = 0; c < 64; ++c) acc += w[j * 64 + c] * f[c];
  h1[b * 128 + j] = acc;
}

__global__ __launch_bounds__(128)
void ldpc_head2(const float* __restrict__ h1, float* __restrict__ mean,
                float* __restrict__ inv) {
  const int j = threadIdx.x;
  float s = 0.f, s2 = 0.f;
  for (int b = 0; b < 512; ++b) {
    const float v = h1[b * 128 + j];
    s += v; s2 += v * v;
  }
  const float m = s * (1.f / 512.f);
  const float var = s2 * (1.f / 512.f) - m * m;
  mean[j] = m;
  inv[j] = rsqrtf(var + 1e-5f);
}

__global__ __launch_bounds__(128)
void ldpc_head3(KP P, const float* __restrict__ h1, const float* __restrict__ mean,
                const float* __restrict__ inv, float* __restrict__ snr) {
  __shared__ float hn[128], h2s[128], red[128];
  const int b = blockIdx.x, j = threadIdx.x;
  float v = (h1[b * 128 + j] - mean[j]) * inv[j] * PF(P_BN_G)[j] + PF(P_BN_B)[j];
  hn[j] = fmaxf(v, 0.f);
  __syncthreads();
  const float* w2 = PF(P_REG_W2);
  float acc = PF(P_REG_B2)[j];
  for (int k = 0; k < 128; ++k) acc += w2[j * 128 + k] * hn[k];
  h2s[j] = fmaxf(acc, 0.f);
  __syncthreads();
  red[j] = PF(P_REG_W3)[j] * h2s[j];
  __syncthreads();
  for (int s = 64; s > 0; s >>= 1) {
    if (j < s) red[j] += red[j + s];
    __syncthreads();
  }
  if (j == 0) snr[b] = fmaxf(red[0] + PF(P_REG_B3)[0], 0.f);
}

// ==================== launch ====================

extern "C" void kernel_launch(void* const* d_in, const int* in_sizes, int n_in,
                              void* d_out, int out_size, void* d_ws, size_t ws_size,
                              hipStream_t stream) {
  (void)in_sizes; (void)out_size; (void)ws_size;
  KP P;
  for (int i = 0; i < 128; ++i) P.p[i] = (i < n_in) ? d_in[i] : nullptr;

  float* bits = (float*)d_out;              // 512*48
  float* snr  = (float*)d_out + 512 * 48;   // 512
  float* f1   = (float*)d_ws;               // 512*64
  float* h1   = f1 + 512 * 64;              // 512*128
  float* mean = h1 + 512 * 128;             // 128
  float* inv  = mean + 128;                 // 128

  ldpc_main<<<512, 256, 0, stream>>>(P, bits, f1);
  ldpc_head1<<<512, 128, 0, stream>>>(P, f1, h1);
  ldpc_head2<<<1, 128, 0, stream>>>(h1, mean, inv);
  ldpc_head3<<<512, 128, 0, stream>>>(P, h1, mean, inv, snr);
}
